// PolymerJEPA_85298050498625
// MI455X (gfx1250) — matrix operand order repaired and checked by reading, not compiled, not run on hardware
//
#include <hip/hip_runtime.h>
#include <hip/hip_bf16.h>

typedef __attribute__((ext_vector_type(16))) __bf16 v16bf;
typedef __attribute__((ext_vector_type(8)))  __bf16 v8bf;
typedef __attribute__((ext_vector_type(8)))  float  v8f;

union V16U { v16bf v; v8bf h[2]; };

constexpr int kNHID = 128, kHWD = 300, kRW = 20, kNFN = 133, kNFE = 14, kNTGT = 4;
constexpr int kNN = 50000, kNS = 150000, kES = 300000;
constexpr int kB = 256, kP = 8, kTP = 2048;

enum { MXIN = 0, MMSG0 = 1, MMSG1 = 2, MUPD0 = 3, MUPD1 = 4, MU0 = 5, MLIN = 6 };

struct GArgs {
  const float* __restrict__ A;     // dense A base, or gather base
  const float* __restrict__ EA;    // edge_attr (MSG)
  const int* __restrict__ rmap;    // snm (XIN) or src (MSG)
  const int* __restrict__ rmap2;   // snm for MSG1 double-gather
  const int* __restrict__ emap;    // sem (MSG)
  const float* __restrict__ W;     // [K,N] f32
  const float* __restrict__ bias;  // [N]
  const float* __restrict__ res1;  // hnode (UPD1, gathered via residx)
  const int* __restrict__ residx;  // snm (UPD*)
  const float* __restrict__ res2;  // xh0 (UPD0/UPD1 direct residual)
  const float* __restrict__ svec;  // node_weight (UPD*) or edge_weight (MSG)
  const int* __restrict__ oidx;    // dst (MSG) / subgraphs_batch (LIN)
  float* __restrict__ OUT;
  int rows, K, N;
};

// ---------------------------------------------------------------------------
// bf16 WMMA GEMM: 256 threads (8 waves), 128x64 block tile, software-pipelined
// global->reg->LDS staging. Wave w owns rows [w*16, w*16+16) x all 64 cols ->
// 4 accumulators, 4 WMMA per A fragment. Tile kt+1's global loads are issued
// before tile kt's WMMAs so HBM latency overlaps the matrix pipe.
// ---------------------------------------------------------------------------
template <int MODE>
__global__ __launch_bounds__(256) void gemm_k(GArgs g) {
  __shared__ __align__(16) unsigned char smA[128 * 80];
  __shared__ __align__(16) unsigned char smB[64 * 80];
  const int tid  = threadIdx.x;
  const int lane = tid & 31;
  const int wave = tid >> 5;  // 0..7: row strip
  const int hi = lane >> 4, lm = lane & 15;
  const int rowBase = blockIdx.x * 128;
  const int colBase = blockIdx.y * 64;

  // A staging: thread -> (row, 16 consecutive k)
  const int sRow = tid >> 1;        // 0..127
  const int sK0  = (tid & 1) * 16;  // 0 or 16
  // B staging: thread -> (col, 8 consecutive k)
  const int sN   = tid & 63;        // 0..63
  const int sKB0 = (tid >> 6) * 8;  // 0,8,16,24

  const int r  = rowBase + sRow;
  const bool rv = r < g.rows;
  const int rc = rv ? r : 0;  // clamp so gather pointers are always valid
  const float* arow;
  const float* earow = g.EA;
  if constexpr (MODE == MXIN) {
    arow = g.A + (size_t)g.rmap[rc] * kNFN;
  } else if constexpr (MODE == MMSG0 || MODE == MMSG1) {
    int i = g.rmap[rc];
    if constexpr (MODE == MMSG1) i = g.rmap2[i];
    arow  = g.A + (size_t)i * kHWD;
    earow = g.EA + (size_t)g.emap[rc] * kNFE;
  } else {
    arow = g.A + (size_t)rc * kHWD;
  }

  const int  nc    = colBase + sN;
  const int  ncc   = nc < g.N ? nc : g.N - 1;
  const bool fullN = (colBase + 64 <= g.N);

  float fa[16];  // in-flight A tile (raw f32)
  float fb[8];   // in-flight B tile (raw f32)

  // fetch: address math + unconditional loads only (no use of loaded values,
  // so the compiler defers s_wait to the store phase of the next iteration)
  auto fetchA = [&](int kb) {
    const float* pbase = arow + kb + sK0;
    bool inK;
    if constexpr (MODE == MMSG0 || MODE == MMSG1) inK = (kb + 32 <= kHWD);
    else inK = (kb + 32 <= g.K);
    if (inK) {
#pragma unroll
      for (int j = 0; j < 16; ++j) fa[j] = pbase[j];
    } else {
#pragma unroll
      for (int j = 0; j < 16; ++j) {
        int kk = kb + sK0 + j;
        const float* p;
        if constexpr (MODE == MMSG0 || MODE == MMSG1) {
          // h-part reads past the row end stay inside the ws allocation;
          // edge_attr part uses a clamped index into the gathered edge row
          int ec = kk - kHWD; ec = ec < 0 ? 0 : (ec > kNFE - 1 ? kNFE - 1 : ec);
          p = (kk < kHWD) ? (pbase + j) : (earow + ec);
        } else if constexpr (MODE == MXIN) {
          // input x boundary: clamp (do not read past the input buffer)
          p = arow + (kk < g.K ? kk : g.K - 1);
        } else {
          // dense ws buffers: past-row reads stay inside the ws allocation
          p = pbase + j;
        }
        fa[j] = *p;
      }
    }
  };
  auto fetchB = [&](int kb) {
    if ((kb + 32 <= g.K) && fullN) {
#pragma unroll
      for (int j = 0; j < 8; ++j) fb[j] = g.W[(size_t)(kb + sKB0 + j) * g.N + nc];
    } else {
#pragma unroll
      for (int j = 0; j < 8; ++j) {
        int kk = kb + sKB0 + j;
        int kc = kk < g.K ? kk : (g.K - 1);
        fb[j] = g.W[(size_t)kc * g.N + ncc];
      }
    }
  };
  // store: validity select + f32->bf16 convert + LDS write
  auto storeA = [&](int kb) {
    bool inK;
    if constexpr (MODE == MMSG0 || MODE == MMSG1) inK = (kb + 32 <= kHWD);
    else inK = (kb + 32 <= g.K);
    v8bf pk0, pk1;
    if (inK) {
#pragma unroll
      for (int j = 0; j < 16; ++j) {
        if (j < 8) pk0[j] = (__bf16)fa[j]; else pk1[j - 8] = (__bf16)fa[j];
      }
    } else {
      const int kEnd = (MODE == MMSG0 || MODE == MMSG1) ? (kHWD + kNFE) : g.K;
#pragma unroll
      for (int j = 0; j < 16; ++j) {
        int kk = kb + sK0 + j;
        float v = (kk < kEnd) ? fa[j] : 0.f;
        if (j < 8) pk0[j] = (__bf16)v; else pk1[j - 8] = (__bf16)v;
      }
    }
    *(v8bf*)(smA + sRow * 80 + sK0 * 2)      = pk0;
    *(v8bf*)(smA + sRow * 80 + sK0 * 2 + 16) = pk1;
  };
  auto storeB = [&](int kb) {
    v8bf pk;
    if ((kb + 32 <= g.K) && fullN) {
#pragma unroll
      for (int j = 0; j < 8; ++j) pk[j] = (__bf16)fb[j];
    } else {
#pragma unroll
      for (int j = 0; j < 8; ++j) {
        int kk = kb + sKB0 + j;
        pk[j] = (__bf16)((kk < g.K && nc < g.N) ? fb[j] : 0.f);
      }
    }
    *(v8bf*)(smB + sN * 80 + sKB0 * 2) = pk;
  };

  v8f acc[4] = {v8f{}, v8f{}, v8f{}, v8f{}};
  const int KT = (g.K + 31) / 32;

  fetchA(0);
  fetchB(0);
  for (int kt = 0; kt < KT; ++kt) {
    const int kb = kt * 32;
    __syncthreads();           // prior iteration's LDS reads done
    storeA(kb);                // waits for in-flight loads land here
    storeB(kb);
    __syncthreads();           // stores visible
    if (kt + 1 < KT) {         // issue next tile's global loads now;
      fetchA(kb + 32);         // they overlap the WMMAs below
      fetchB(kb + 32);
    }

    V16U a;
    const unsigned char* pa = smA + (wave * 16 + lm) * 80 + hi * 16;
    a.h[0] = *(const v8bf*)(pa);
    a.h[1] = *(const v8bf*)(pa + 32);
    V16U b[4];
#pragma unroll
    for (int c = 0; c < 4; ++c) {
      const unsigned char* pb = smB + (c * 16 + lm) * 80 + hi * 16;
      b[c].h[0] = *(const v8bf*)(pb);
      b[c].h[1] = *(const v8bf*)(pb + 32);
    }
#pragma unroll
    for (int c = 0; c < 4; ++c) {
      acc[c] = __builtin_amdgcn_wmma_f32_16x16x32_bf16(false, a.v, false, b[c].v, (short)0, acc[c], false, false);
    }
  }

  // ---- epilogue: lane holds D[(hi*8+v), lm] of each 16x16 tile ----
#pragma unroll
  for (int c = 0; c < 4; ++c) {
    int n = colBase + c * 16 + lm;
    if (n >= g.N) continue;
    float bia = g.bias[n];
#pragma unroll
    for (int vv = 0; vv < 8; ++vv) {
      int m = rowBase + wave * 16 + hi * 8 + vv;
      if (m >= g.rows) continue;
      float val = acc[c][vv] + bia;
      if constexpr (MODE == MXIN || MODE == MU0) {
        g.OUT[(size_t)m * g.N + n] = fmaxf(val, 0.f);
      } else if constexpr (MODE == MMSG0 || MODE == MMSG1) {
        float ew = g.svec[g.emap[m]];
        atomicAdd(&g.OUT[(size_t)g.oidx[m] * kHWD + n], fmaxf(val, 0.f) * ew);
      } else if constexpr (MODE == MUPD0) {
        float nw = g.svec[g.residx[m]];
        g.OUT[(size_t)m * kHWD + n] = fmaxf(g.res2[(size_t)m * kHWD + n] + val, 0.f) * nw;
      } else if constexpr (MODE == MUPD1) {
        int si = g.residx[m];
        float nw = g.svec[si];
        float r1 = g.res1[(size_t)si * kHWD + n];
        g.OUT[(size_t)m * kHWD + n] = (fmaxf(r1 + val, 0.f) + g.res2[(size_t)m * kHWD + n]) * nw;
      } else if constexpr (MODE == MLIN) {
        atomicAdd(&g.OUT[(size_t)g.oidx[m] * kNHID + n], val);
      }
    }
  }
}

// ---------------------------------------------------------------------------
// segment / elementwise helper kernels
// ---------------------------------------------------------------------------
__global__ void count_k(const int* sb, const int* snm, float* cnt_tp, float* cnt_node, int n) {
  int i = blockIdx.x * 256 + threadIdx.x;
  if (i < n) {
    atomicAdd(&cnt_tp[sb[i]], 1.f);
    atomicAdd(&cnt_node[snm[i]], 1.f);
  }
}

__global__ void segmax_k(const float* rwpe, const int* snm, const int* sb, unsigned* penc, int n) {
  int i = blockIdx.x * 256 + threadIdx.x;
  if (i < n) {
    int rr = i / kRW, cc = i % kRW;
    float f = rwpe[(size_t)snm[rr] * kRW + cc];
    unsigned u = __float_as_uint(f);
    unsigned kk = (u >> 31) ? ~u : (u | 0x80000000u);  // order-preserving encode
    atomicMax(&penc[(size_t)sb[rr] * kRW + cc], kk);
  }
}

__global__ void decode_k(const unsigned* penc, float* ppes, int n) {
  int i = blockIdx.x * 256 + threadIdx.x;
  if (i < n) {
    unsigned kk = penc[i];
    unsigned u = (kk >> 31) ? (kk & 0x7fffffffu) : ~kk;
    float f = __uint_as_float(u);
    if (!(f == f) || fabsf(f) > 3.0e38f) f = 0.f;  // empty/non-finite -> 0
    ppes[i] = f;
  }
}

__global__ void segsum_k(const float* src, const int* ids, float* dst, int rows, int cols) {
  int i = blockIdx.x * 256 + threadIdx.x;
  if (i < rows * cols) {
    int rr = i / cols, cc = i % cols;
    atomicAdd(&dst[(size_t)ids[rr] * cols + cc], src[i]);
  }
}

__global__ void divmean_k(float* buf, const float* cnt, int rows, int cols) {
  int i = blockIdx.x * 256 + threadIdx.x;
  if (i < rows * cols) buf[i] /= fmaxf(cnt[i / cols], 1.f);
}

__global__ void addgather_k(float* h, const float* subU, const int* sb, int n) {
  int i = blockIdx.x * 256 + threadIdx.x;
  if (i < n) {
    int rr = i / kHWD, cc = i % kHWD;
    h[i] += subU[(size_t)sb[rr] * kHWD + cc];
  }
}

__global__ __launch_bounds__(128) void ctxin_k(const float* sgx, const float* ppes,
                                               const float* rwW, const float* rwb,
                                               const int* cidx, float* out) {
  int b = blockIdx.x, t = threadIdx.x;
  int ci = cidx[b] + b * kP;
  float a = rwb[t];
  for (int j = 0; j < kRW; ++j) a += ppes[(size_t)ci * kRW + j] * rwW[j * kNHID + t];
  out[(size_t)b * kNHID + t] = sgx[(size_t)ci * kNHID + t] + a;
}

// ---------------------------------------------------------------------------
// small transformer kernels (S<=8, D=128, 4 heads x 32)
// ---------------------------------------------------------------------------
struct EncP {
  const float *Wq, *bq, *Wk, *bk, *Wv, *bv, *Wo, *bo;
  const float *g1, *be1, *W1, *b1, *W2, *b2, *g2, *be2;
};

__global__ __launch_bounds__(128) void ctx_enc_k(const float* xin, EncP p0, EncP p1, float* xout) {
  __shared__ float xs[128], vv[128], yy[128], ff[256], st[2];
  int b = blockIdx.x, t = threadIdx.x;
  xs[t] = xin[(size_t)b * 128 + t];
  __syncthreads();
  EncP ps[2] = {p0, p1};
  for (int L = 0; L < 2; ++L) {
    const EncP p = ps[L];
    // S=1 attention: softmax over one key == 1, so o = v
    float a = p.bv[t];
    for (int k = 0; k < 128; ++k) a += xs[k] * p.Wv[k * 128 + t];
    vv[t] = a;
    __syncthreads();
    a = p.bo[t];
    for (int k = 0; k < 128; ++k) a += vv[k] * p.Wo[k * 128 + t];
    yy[t] = xs[t] + a;
    __syncthreads();
    if (t == 0) {
      float m = 0.f; for (int k = 0; k < 128; ++k) m += yy[k]; m *= (1.f / 128.f);
      float v2 = 0.f; for (int k = 0; k < 128; ++k) { float d = yy[k] - m; v2 += d * d; }
      st[0] = m; st[1] = rsqrtf(v2 * (1.f / 128.f) + 1e-5f);
    }
    __syncthreads();
    xs[t] = (yy[t] - st[0]) * st[1] * p.g1[t] + p.be1[t];
    __syncthreads();
    for (int e = t; e < 256; e += 128) {
      float aa = p.b1[e];
      for (int k = 0; k < 128; ++k) aa += xs[k] * p.W1[k * 256 + e];
      ff[e] = fmaxf(aa, 0.f);
    }
    __syncthreads();
    a = p.b2[t];
    for (int j = 0; j < 256; ++j) a += ff[j] * p.W2[j * 128 + t];
    yy[t] = xs[t] + a;
    __syncthreads();
    if (t == 0) {
      float m = 0.f; for (int k = 0; k < 128; ++k) m += yy[k]; m *= (1.f / 128.f);
      float v2 = 0.f; for (int k = 0; k < 128; ++k) { float d = yy[k] - m; v2 += d * d; }
      st[0] = m; st[1] = rsqrtf(v2 * (1.f / 128.f) + 1e-5f);
    }
    __syncthreads();
    xs[t] = (yy[t] - st[0]) * st[1] * p.g2[t] + p.be2[t];
    __syncthreads();
  }
  xout[(size_t)b * 128 + t] = xs[t];
}

__global__ __launch_bounds__(128) void tgt_enc_k(const float* sgx, const int* mask,
                                                 EncP p0, EncP p1, float* mixer) {
  __shared__ float xs[8][128], qs[8][128], ks2[8][128], vs[8][128];
  __shared__ float att[4][8][8], ffn[8][256], st[8][2], padd[8];
  int b = blockIdx.x, t = threadIdx.x;
  for (int s = 0; s < 8; ++s) xs[s][t] = sgx[(size_t)(b * 8 + s) * 128 + t];
  if (t < 8) padd[t] = mask[b * 8 + t] ? 0.f : -1e9f;
  __syncthreads();
  EncP ps[2] = {p0, p1};
  for (int L = 0; L < 2; ++L) {
    const EncP p = ps[L];
    for (int s = 0; s < 8; ++s) {
      float aq = p.bq[t], ak = p.bk[t], av = p.bv[t];
      for (int k = 0; k < 128; ++k) {
        float xv = xs[s][k];
        aq += xv * p.Wq[k * 128 + t];
        ak += xv * p.Wk[k * 128 + t];
        av += xv * p.Wv[k * 128 + t];
      }
      qs[s][t] = aq; ks2[s][t] = ak; vs[s][t] = av;
    }
    __syncthreads();
    for (int e = t; e < 256; e += 128) {
      int hh = e >> 6, rem = e & 63, qi = rem >> 3, kj = rem & 7;
      float sc = 0.f;
      for (int d = 0; d < 32; ++d) sc += qs[qi][hh * 32 + d] * ks2[kj][hh * 32 + d];
      att[hh][qi][kj] = sc * 0.17677669529663687f + padd[kj];
    }
    __syncthreads();
    if (t < 32) {
      int hh = t >> 3, qi = t & 7;
      float mx = -1e30f;
      for (int j = 0; j < 8; ++j) mx = fmaxf(mx, att[hh][qi][j]);
      float sm = 0.f;
      for (int j = 0; j < 8; ++j) { float e2 = __expf(att[hh][qi][j] - mx); att[hh][qi][j] = e2; sm += e2; }
      float inv = 1.f / sm;
      for (int j = 0; j < 8; ++j) att[hh][qi][j] *= inv;
    }
    __syncthreads();
    {
      int hh = t >> 5;
      float o[8];
      for (int s = 0; s < 8; ++s) {
        float a2 = 0.f;
        for (int j = 0; j < 8; ++j) a2 += att[hh][s][j] * vs[j][t];
        o[s] = a2;
      }
      __syncthreads();
      for (int s = 0; s < 8; ++s) qs[s][t] = o[s];
    }
    __syncthreads();
    for (int s = 0; s < 8; ++s) {
      float a2 = p.bo[t];
      for (int k = 0; k < 128; ++k) a2 += qs[s][k] * p.Wo[k * 128 + t];
      ks2[s][t] = xs[s][t] + a2;
    }
    __syncthreads();
    if (t < 8) {
      float m = 0.f; for (int k = 0; k < 128; ++k) m += ks2[t][k]; m *= (1.f / 128.f);
      float v2 = 0.f; for (int k = 0; k < 128; ++k) { float d = ks2[t][k] - m; v2 += d * d; }
      st[t][0] = m; st[t][1] = rsqrtf(v2 * (1.f / 128.f) + 1e-5f);
    }
    __syncthreads();
    for (int s = 0; s < 8; ++s) xs[s][t] = (ks2[s][t] - st[s][0]) * st[s][1] * p.g1[t] + p.be1[t];
    __syncthreads();
    for (int e = t; e < 256; e += 128) {
      for (int s = 0; s < 8; ++s) {
        float a2 = p.b1[e];
        for (int k = 0; k < 128; ++k) a2 += xs[s][k] * p.W1[k * 256 + e];
        ffn[s][e] = fmaxf(a2, 0.f);
      }
    }
    __syncthreads();
    for (int s = 0; s < 8; ++s) {
      float a2 = p.b2[t];
      for (int j = 0; j < 256; ++j) a2 += ffn[s][j] * p.W2[j * 128 + t];
      ks2[s][t] = xs[s][t] + a2;
    }
    __syncthreads();
    if (t < 8) {
      float m = 0.f; for (int k = 0; k < 128; ++k) m += ks2[t][k]; m *= (1.f / 128.f);
      float v2 = 0.f; for (int k = 0; k < 128; ++k) { float d = ks2[t][k] - m; v2 += d * d; }
      st[t][0] = m; st[t][1] = rsqrtf(v2 * (1.f / 128.f) + 1e-5f);
    }
    __syncthreads();
    for (int s = 0; s < 8; ++s) xs[s][t] = (ks2[s][t] - st[s][0]) * st[s][1] * p.g2[t] + p.be2[t];
    __syncthreads();
  }
  for (int s = 0; s < 8; ++s) mixer[(size_t)(b * 8 + s) * 128 + t] = xs[s][t];
}

__global__ __launch_bounds__(128) void pred_k(const float* mixer, const float* ctx_emb,
                                              const float* ppes, const float* rwW, const float* rwb,
                                              const int* tgtIdx,
                                              const float* W1, const float* b1,
                                              const float* lg, const float* lb,
                                              const float* W2, const float* b2, float* out) {
  __shared__ float pin[128], yy[128], zz[128], st[2];
  int blk = blockIdx.x, b = blk >> 2, tt = blk & 3, t = threadIdx.x;
  int ti = tgtIdx[b * kNTGT + tt] + b * kP;
  out[(size_t)blk * 128 + t] = mixer[(size_t)ti * 128 + t];  // target_emb
  float a = rwb[t];
  for (int j = 0; j < kRW; ++j) a += ppes[(size_t)ti * kRW + j] * rwW[j * 128 + t];
  pin[t] = ctx_emb[(size_t)b * 128 + t] + a;
  __syncthreads();
  a = b1[t];
  for (int k = 0; k < 128; ++k) a += pin[k] * W1[k * 128 + t];
  yy[t] = a;
  __syncthreads();
  if (t == 0) {
    float m = 0.f; for (int k = 0; k < 128; ++k) m += yy[k]; m *= (1.f / 128.f);
    float v2 = 0.f; for (int k = 0; k < 128; ++k) { float d = yy[k] - m; v2 += d * d; }
    st[0] = m; st[1] = rsqrtf(v2 * (1.f / 128.f) + 1e-5f);
  }
  __syncthreads();
  zz[t] = fmaxf((yy[t] - st[0]) * st[1] * lg[t] + lb[t], 0.f);
  __syncthreads();
  a = b2[t];
  for (int k = 0; k < 128; ++k) a += zz[k] * W2[k * 128 + t];
  out[(size_t)kB * kNTGT * 128 + (size_t)blk * 128 + t] = a;  // pred
}

// ---------------------------------------------------------------------------
static EncP mkEnc(void* const* d_in, int base) {
  EncP p;
  const float** f = (const float**)&p;
  for (int i = 0; i < 16; ++i) f[i] = (const float*)d_in[base + i];
  return p;
}

extern "C" void kernel_launch(void* const* d_in, const int* in_sizes, int n_in,
                              void* d_out, int out_size, void* d_ws, size_t ws_size,
                              hipStream_t stream) {
  (void)in_sizes; (void)n_in; (void)out_size; (void)ws_size;
  const float* x      = (const float*)d_in[0];
  const float* node_w = (const float*)d_in[1];
  const float* eattr  = (const float*)d_in[2];
  const float* ewt    = (const float*)d_in[3];
  const float* rwpe   = (const float*)d_in[4];
  const int* snm  = (const int*)d_in[5];
  const int* sem  = (const int*)d_in[6];
  const int* src  = (const int*)d_in[7];
  const int* dst  = src + kES;
  const int* sb   = (const int*)d_in[8];
  const int* mskp = (const int*)d_in[9];
  const int* cidx = (const int*)d_in[10];
  const int* tidx = (const int*)d_in[11];
  auto F = [&](int i) { return (const float*)d_in[i]; };

  // workspace layout (floats; every offset is a multiple of 16 floats -> 64B)
  float* ws = (float*)d_ws;
  size_t o = 0;
  float* xh0      = ws + o; o += (size_t)kNS * kHWD;
  float* hbuf     = ws + o; o += (size_t)kNS * kHWD;
  float* agg      = ws + o; o += (size_t)kNS * kHWD;
  float* hnode    = ws + o; o += (size_t)kNN * kHWD;
  float* cnt_node = ws + o; o += kNN;
  float* sub_tp   = ws + o; o += (size_t)kTP * kHWD;
  float* cnt_tp   = ws + o; o += kTP;
  float* subU     = ws + o; o += (size_t)kTP * kHWD;
  unsigned* penc  = (unsigned*)(ws + o); o += (size_t)kTP * kRW;
  float* ppes     = ws + o; o += (size_t)kTP * kRW;
  float* sgx      = ws + o; o += (size_t)kTP * kNHID;
  float* ctx_in   = ws + o; o += (size_t)kB * kNHID;
  float* ctx_emb  = ws + o; o += (size_t)kB * kNHID;
  float* mixer    = ws + o; o += (size_t)kTP * kNHID;

  // zero all accumulators (deterministic per call)
  hipMemsetAsync(agg, 0, (size_t)kNS * kHWD * 4, stream);
  hipMemsetAsync(hnode, 0, (size_t)kNN * kHWD * 4, stream);
  hipMemsetAsync(cnt_node, 0, (size_t)kNN * 4, stream);
  hipMemsetAsync(sub_tp, 0, (size_t)kTP * kHWD * 4, stream);
  hipMemsetAsync(cnt_tp, 0, (size_t)kTP * 4, stream);
  hipMemsetAsync(penc, 0, (size_t)kTP * kRW * 4, stream);
  hipMemsetAsync(sgx, 0, (size_t)kTP * kNHID * 4, stream);

  count_k<<<(kNS + 255) / 256, 256, 0, stream>>>(sb, snm, cnt_tp, cnt_node, kNS);
  segmax_k<<<(kNS * kRW + 255) / 256, 256, 0, stream>>>(rwpe, snm, sb, penc, kNS * kRW);
  decode_k<<<(kTP * kRW + 255) / 256, 256, 0, stream>>>(penc, ppes, kTP * kRW);

  // xh0 = relu(x[snm] @ W_in + b_in)
  { GArgs g{}; g.A = x; g.rmap = snm; g.W = F(13); g.bias = F(14); g.OUT = xh0;
    g.rows = kNS; g.K = kNFN; g.N = kHWD;
    gemm_k<MXIN><<<dim3((kNS + 127) / 128, 5), 256, 0, stream>>>(g); }
  // msg0: relu([xh0[src]|ea] @ Wm + bm)*ew -> atomic scatter agg[dst]
  { GArgs g{}; g.A = xh0; g.EA = eattr; g.rmap = src; g.emap = sem;
    g.W = F(15); g.bias = F(16); g.svec = ewt; g.oidx = dst; g.OUT = agg;
    g.rows = kES; g.K = kHWD + kNFE; g.N = kHWD;
    gemm_k<MMSG0><<<dim3((kES + 127) / 128, 5), 256, 0, stream>>>(g); }
  // h = relu(xh0 + agg@Wu + bu) * node_weight[snm]
  { GArgs g{}; g.A = agg; g.W = F(17); g.bias = F(18); g.res2 = xh0;
    g.residx = snm; g.svec = node_w; g.OUT = hbuf;
    g.rows = kNS; g.K = kHWD; g.N = kHWD;
    gemm_k<MUPD0><<<dim3((kNS + 127) / 128, 5), 256, 0, stream>>>(g); }
  // sub = seg_mean(h, sb, TP); subU = relu(sub@U0+b); h += subU[sb]
  segsum_k<<<(kNS * kHWD + 255) / 256, 256, 0, stream>>>(hbuf, sb, sub_tp, kNS, kHWD);
  divmean_k<<<(kTP * kHWD + 255) / 256, 256, 0, stream>>>(sub_tp, cnt_tp, kTP, kHWD);
  { GArgs g{}; g.A = sub_tp; g.W = F(23); g.bias = F(24); g.OUT = subU;
    g.rows = kTP; g.K = kHWD; g.N = kHWD;
    gemm_k<MU0><<<dim3(kTP / 128, 5), 256, 0, stream>>>(g); }
  addgather_k<<<(kNS * kHWD + 255) / 256, 256, 0, stream>>>(hbuf, subU, sb, kNS * kHWD);
  // hnode = seg_mean(h, snm, N_NODES)
  segsum_k<<<(kNS * kHWD + 255) / 256, 256, 0, stream>>>(hbuf, snm, hnode, kNS, kHWD);
  divmean_k<<<(kNN * kHWD + 255) / 256, 256, 0, stream>>>(hnode, cnt_node, kNN, kHWD);
  // msg1 with double gather hnode[snm[src]]
  hipMemsetAsync(agg, 0, (size_t)kNS * kHWD * 4, stream);
  { GArgs g{}; g.A = hnode; g.EA = eattr; g.rmap = src; g.rmap2 = snm; g.emap = sem;
    g.W = F(19); g.bias = F(20); g.svec = ewt; g.oidx = dst; g.OUT = agg;
    g.rows = kES; g.K = kHWD + kNFE; g.N = kHWD;
    gemm_k<MMSG1><<<dim3((kES + 127) / 128, 5), 256, 0, stream>>>(g); }
  // h = (relu(hnode[snm] + agg@Wu + bu) + xh0) * node_weight[snm]
  { GArgs g{}; g.A = agg; g.W = F(21); g.bias = F(22); g.res1 = hnode;
    g.residx = snm; g.res2 = xh0; g.svec = node_w; g.OUT = hbuf;
    g.rows = kNS; g.K = kHWD; g.N = kHWD;
    gemm_k<MUPD1><<<dim3((kNS + 127) / 128, 5), 256, 0, stream>>>(g); }
  // lin fused with segment-sum into sgx; then /counts  (N=128 -> 2 col blocks)
  { GArgs g{}; g.A = hbuf; g.W = F(25); g.bias = F(26); g.oidx = sb; g.OUT = sgx;
    g.rows = kNS; g.K = kHWD; g.N = kNHID;
    gemm_k<MLIN><<<dim3((kNS + 127) / 128, 2), 256, 0, stream>>>(g); }
  divmean_k<<<(kTP * kNHID + 255) / 256, 256, 0, stream>>>(sgx, cnt_tp, kTP, kNHID);

  // transformer heads
  ctxin_k<<<kB, 128, 0, stream>>>(sgx, ppes, F(27), F(28), cidx, ctx_in);
  ctx_enc_k<<<kB, 128, 0, stream>>>(ctx_in, mkEnc(d_in, 29), mkEnc(d_in, 45), ctx_emb);
  tgt_enc_k<<<kB, 128, 0, stream>>>(sgx, mskp, mkEnc(d_in, 61), mkEnc(d_in, 77), mixer);
  pred_k<<<kB * kNTGT, 128, 0, stream>>>(mixer, ctx_emb, ppes, F(27), F(28), tidx,
                                         F(93), F(94), F(95), F(96), F(97), F(98),
                                         (float*)d_out);
}
